// MQA_4475355922804
// MI455X (gfx1250) — matrix operand order repaired and checked by reading, not compiled, not run on hardware
//
#include <hip/hip_runtime.h>
#include <hip/hip_bf16.h>

#define S_DIM 2048
#define HID   2048
#define NHEAD 16
#define HDIM  128
#define SW    (S_DIM / 32)   // mask words per row

typedef __attribute__((ext_vector_type(16))) __bf16 v16bf;
typedef __attribute__((ext_vector_type(8)))  __bf16 v8bf;
typedef __attribute__((ext_vector_type(8)))  float  v8f;
typedef __attribute__((ext_vector_type(4)))  float  v4f;

static __device__ __forceinline__ v16bf join16(v8bf lo, v8bf hi) {
  v16bf r;
#pragma unroll
  for (int i = 0; i < 8; ++i) { r[i] = lo[i]; r[i + 8] = hi[i]; }
  return r;
}

// ---------------------------------------------------------------- convert ---
__global__ void __launch_bounds__(256)
f32_to_bf16_kernel(const float* __restrict__ src, __bf16* __restrict__ dst, int n) {
  int i = (blockIdx.x * 256 + threadIdx.x) * 8;
  if (i >= n) return;
  v4f a = *(const v4f*)(src + i);
  v4f b = *(const v4f*)(src + i + 4);
  v8bf o;
#pragma unroll
  for (int j = 0; j < 4; ++j) { o[j] = (__bf16)a[j]; o[j + 4] = (__bf16)b[j]; }
  *(v8bf*)(dst + i) = o;
}

// -------------------------------------------------------------- mask pack ---
// (B,S,S) int32 -> bitmask, one u32 per 32 keys (wave32 ballot).
__global__ void __launch_bounds__(256)
pack_mask_kernel(const int* __restrict__ mask, unsigned* __restrict__ bits) {
  int i = blockIdx.x * 256 + threadIdx.x;
  unsigned long long bal = __ballot(mask[i] != 0);
  if ((threadIdx.x & 31) == 0) bits[i >> 5] = (unsigned)bal;
}

// ------------------------------------------------------------------- GEMM ---
// C[M,N] = A[M,K](bf16,row) @ B[K,N](bf16,row) + bias[N]; 256x128 tile/block,
// 8 waves, wave = 32 rows x 128 cols (2x8 WMMA accumulators; B-frag reused 2x).
template <typename OutT>
__global__ void __launch_bounds__(256)
gemm_bf16_wmma(const __bf16* __restrict__ A, const __bf16* __restrict__ Bm,
               const float* __restrict__ bias, OutT* __restrict__ C,
               int M, int N, int K) {
  __shared__ __align__(64) __bf16 ldsA[256][32];   // [m][k]
  __shared__ __align__(64) __bf16 ldsBT[128][32];  // [n][k] (transposed)
  const int tid = threadIdx.x;
  const int lane = tid & 31, wv = tid >> 5;
  const int hlf = lane >> 4, ln = lane & 15;
  const int m0 = blockIdx.y * 256, n0 = blockIdx.x * 128;

  v8f acc[2][8];
#pragma unroll
  for (int nt = 0; nt < 8; ++nt) {
    float bv = bias[n0 + nt * 16 + ln];
#pragma unroll
    for (int r = 0; r < 8; ++r) { acc[0][nt][r] = bv; acc[1][nt][r] = bv; }
  }

  const int bk = tid >> 3, bn = (tid & 7) * 16;  // B staging: 8 thr/row

  for (int k0 = 0; k0 < K; k0 += 32) {
    {  // A staging: 1 thread = 1 row of 32 (64B contiguous)
      const __bf16* s = A + (size_t)(m0 + tid) * K + k0;
      *(v8bf*)&ldsA[tid][0]  = *(const v8bf*)s;
      *(v8bf*)&ldsA[tid][8]  = *(const v8bf*)(s + 8);
      *(v8bf*)&ldsA[tid][16] = *(const v8bf*)(s + 16);
      *(v8bf*)&ldsA[tid][24] = *(const v8bf*)(s + 24);
    }
    {  // B staging, transposed
      const __bf16* s = Bm + (size_t)(k0 + bk) * N + n0 + bn;
      v8bf b0 = *(const v8bf*)s;
      v8bf b1 = *(const v8bf*)(s + 8);
#pragma unroll
      for (int i = 0; i < 8; ++i) {
        ldsBT[bn + i][bk] = b0[i];
        ldsBT[bn + 8 + i][bk] = b1[i];
      }
    }
    if (k0 + 32 < K) {  // global_prefetch_b8 for next K-slab
      __builtin_prefetch(A + (size_t)(m0 + tid) * K + k0 + 32, 0, 1);
      __builtin_prefetch(Bm + (size_t)(k0 + 32 + bk) * N + n0 + bn, 0, 1);
    }
    __syncthreads();

    const int ar0 = wv * 32 + ln;  // A-frag: M = lane%16
    v16bf af0 = join16(*(const v8bf*)&ldsA[ar0][hlf * 8],
                       *(const v8bf*)&ldsA[ar0][16 + hlf * 8]);
    v16bf af1 = join16(*(const v8bf*)&ldsA[ar0 + 16][hlf * 8],
                       *(const v8bf*)&ldsA[ar0 + 16][16 + hlf * 8]);
#pragma unroll
    for (int nt = 0; nt < 8; ++nt) {
      v16bf bfrag = join16(*(const v8bf*)&ldsBT[nt * 16 + ln][hlf * 8],
                           *(const v8bf*)&ldsBT[nt * 16 + ln][16 + hlf * 8]);
      acc[0][nt] = __builtin_amdgcn_wmma_f32_16x16x32_bf16(
          false, af0, false, bfrag, (short)0, acc[0][nt], false, false);
      acc[1][nt] = __builtin_amdgcn_wmma_f32_16x16x32_bf16(
          false, af1, false, bfrag, (short)0, acc[1][nt], false, false);
    }
    __syncthreads();
  }

#pragma unroll
  for (int mt = 0; mt < 2; ++mt)
#pragma unroll
    for (int nt = 0; nt < 8; ++nt)
#pragma unroll
      for (int r = 0; r < 8; ++r) {  // D-layout: M = hlf*8+r, N = lane%16
        size_t row = (size_t)(m0 + wv * 32 + mt * 16 + hlf * 8 + r);
        C[row * N + n0 + nt * 16 + ln] = (OutT)acc[mt][nt][r];
      }
}

// -------------------------------------------------------- flash attention ---
// Block = (b, h, 128 query rows); 8 waves, each one 16-row q-tile.
// K/V chunks (32 keys) staged in LDS, shared by all 8 q-tiles (MQA).
__global__ void __launch_bounds__(256)
mqa_flash_attn(const __bf16* __restrict__ Q, const __bf16* __restrict__ Kh,
               const __bf16* __restrict__ Vh, const unsigned* __restrict__ mbits,
               __bf16* __restrict__ O) {
  __shared__ __align__(64) __bf16 ldsK[32][HDIM];   // [key][d]
  __shared__ __align__(64) __bf16 ldsVT[HDIM][32];  // [d][key]
  __shared__ __align__(64) __bf16 ldsP[8][16][32];  // wave-private P tiles
  const int tid = threadIdx.x;
  const int lane = tid & 31, wv = tid >> 5;
  const int hlf = lane >> 4, ln = lane & 15;
  const int b = blockIdx.z, h = blockIdx.y;
  const int q0 = blockIdx.x * 128 + wv * 16;
  const float scale = 0.088388347648318447f;  // 1/sqrt(128)

  v16bf qf[4];  // Q rows in registers: 16 x 128
  {
    const __bf16* qp = Q + (size_t)(b * S_DIM + q0 + ln) * HID + h * HDIM;
#pragma unroll
    for (int c = 0; c < 4; ++c)
      qf[c] = join16(*(const v8bf*)(qp + c * 32 + hlf * 8),
                     *(const v8bf*)(qp + c * 32 + 16 + hlf * 8));
  }

  float mrow[8], lrow[8];
  v8f oacc[8];
#pragma unroll
  for (int r = 0; r < 8; ++r) { mrow[r] = -1e30f; lrow[r] = 0.f; }
#pragma unroll
  for (int nt = 0; nt < 8; ++nt)
#pragma unroll
    for (int r = 0; r < 8; ++r) oacc[nt][r] = 0.f;

  const int sr = tid >> 3, sc = (tid & 7) * 16;  // staging coords
  const size_t kvbase = (size_t)b * S_DIM;
  const size_t mwbase = ((size_t)b * S_DIM + q0 + hlf * 8) * SW;  // bit rows

  for (int kc = 0; kc < S_DIM; kc += 32) {
    {  // stage K as-is, V transposed
      const __bf16* s = Kh + (kvbase + kc + sr) * HDIM + sc;
      *(v8bf*)&ldsK[sr][sc]     = *(const v8bf*)s;
      *(v8bf*)&ldsK[sr][sc + 8] = *(const v8bf*)(s + 8);
      const __bf16* v = Vh + (kvbase + kc + sr) * HDIM + sc;
      v8bf v0 = *(const v8bf*)v;
      v8bf v1 = *(const v8bf*)(v + 8);
#pragma unroll
      for (int i = 0; i < 8; ++i) {
        ldsVT[sc + i][sr] = v0[i];
        ldsVT[sc + 8 + i][sr] = v1[i];
      }
    }
    __syncthreads();

    v8f s0, s1;  // two 16x16 score tiles (keys kc..+15, kc+16..+31)
#pragma unroll
    for (int r = 0; r < 8; ++r) { s0[r] = 0.f; s1[r] = 0.f; }
#pragma unroll
    for (int c = 0; c < 4; ++c) {  // B-frag of K^T = contiguous K-row bytes
      v16bf kf0 = join16(*(const v8bf*)&ldsK[ln][c * 32 + hlf * 8],
                         *(const v8bf*)&ldsK[ln][c * 32 + 16 + hlf * 8]);
      v16bf kf1 = join16(*(const v8bf*)&ldsK[16 + ln][c * 32 + hlf * 8],
                         *(const v8bf*)&ldsK[16 + ln][c * 32 + 16 + hlf * 8]);
      s0 = __builtin_amdgcn_wmma_f32_16x16x32_bf16(false, qf[c], false, kf0,
                                                   (short)0, s0, false, false);
      s1 = __builtin_amdgcn_wmma_f32_16x16x32_bf16(false, qf[c], false, kf1,
                                                   (short)0, s1, false, false);
    }

    float cm[8];
#pragma unroll
    for (int r = 0; r < 8; ++r) {  // scale + bit-packed mask
      unsigned word = mbits[mwbase + (size_t)r * SW + (kc >> 5)];
      s0[r] = (word >> ln) & 1u        ? s0[r] * scale : -1e30f;
      s1[r] = (word >> (16 + ln)) & 1u ? s1[r] * scale : -1e30f;
      cm[r] = fmaxf(s0[r], s1[r]);
    }
#pragma unroll
    for (int off = 8; off >= 1; off >>= 1)  // reduce over 16-lane half-group
#pragma unroll
      for (int r = 0; r < 8; ++r) cm[r] = fmaxf(cm[r], __shfl_xor(cm[r], off, 32));

    float al[8], rs[8];
#pragma unroll
    for (int r = 0; r < 8; ++r) {
      float mn = fmaxf(mrow[r], cm[r]);
      al[r] = __expf(mrow[r] - mn);
      float p0 = __expf(s0[r] - mn);
      float p1 = __expf(s1[r] - mn);
      s0[r] = p0; s1[r] = p1;
      rs[r] = p0 + p1;
      mrow[r] = mn;
    }
#pragma unroll
    for (int off = 8; off >= 1; off >>= 1)
#pragma unroll
      for (int r = 0; r < 8; ++r) rs[r] += __shfl_xor(rs[r], off, 32);
#pragma unroll
    for (int r = 0; r < 8; ++r) lrow[r] = lrow[r] * al[r] + rs[r];
#pragma unroll
    for (int nt = 0; nt < 8; ++nt)
#pragma unroll
      for (int r = 0; r < 8; ++r) oacc[nt][r] *= al[r];

    // D-layout -> A-layout transpose of P via wave-private LDS
#pragma unroll
    for (int r = 0; r < 8; ++r) {
      ldsP[wv][hlf * 8 + r][ln]      = (__bf16)s0[r];
      ldsP[wv][hlf * 8 + r][16 + ln] = (__bf16)s1[r];
    }
    asm volatile("s_wait_dscnt 0" ::: "memory");
    v16bf pf = join16(*(const v8bf*)&ldsP[wv][ln][hlf * 8],
                      *(const v8bf*)&ldsP[wv][ln][16 + hlf * 8]);
#pragma unroll
    for (int nt = 0; nt < 8; ++nt) {  // O += P @ V
      v16bf vf = join16(*(const v8bf*)&ldsVT[nt * 16 + ln][hlf * 8],
                        *(const v8bf*)&ldsVT[nt * 16 + ln][16 + hlf * 8]);
      oacc[nt] = __builtin_amdgcn_wmma_f32_16x16x32_bf16(
          false, pf, false, vf, (short)0, oacc[nt], false, false);
    }
    __syncthreads();
  }

  float inv[8];
#pragma unroll
  for (int r = 0; r < 8; ++r) inv[r] = 1.0f / lrow[r];
#pragma unroll
  for (int nt = 0; nt < 8; ++nt)
#pragma unroll
    for (int r = 0; r < 8; ++r) {
      size_t row = (size_t)b * S_DIM + q0 + hlf * 8 + r;
      O[row * HID + h * HDIM + nt * 16 + ln] = (__bf16)(oacc[nt][r] * inv[r]);
    }
}

// ------------------------------------------------------------------- host ---
extern "C" void kernel_launch(void* const* d_in, const int* in_sizes, int n_in,
                              void* d_out, int out_size, void* d_ws, size_t ws_size,
                              hipStream_t stream) {
  (void)in_sizes; (void)n_in; (void)out_size; (void)ws_size;
  const float* X  = (const float*)d_in[0];
  const int*   mk = (const int*)d_in[1];
  const float* Wq = (const float*)d_in[2];
  const float* bq = (const float*)d_in[3];
  const float* Wk = (const float*)d_in[4];
  const float* bk = (const float*)d_in[5];
  const float* Wv = (const float*)d_in[6];
  const float* bv = (const float*)d_in[7];
  const float* Wo = (const float*)d_in[8];
  const float* bo = (const float*)d_in[9];
  float* out = (float*)d_out;

  const int BS = 2 * S_DIM;  // 4096 token rows
  char* w = (char*)d_ws;
  __bf16* Xbf = (__bf16*)w; w += (size_t)BS * HID * 2;
  __bf16* Wqb = (__bf16*)w; w += (size_t)HID * HID * 2;
  __bf16* Wkb = (__bf16*)w; w += (size_t)HID * HDIM * 2;
  __bf16* Wvb = (__bf16*)w; w += (size_t)HID * HDIM * 2;
  __bf16* Wob = (__bf16*)w; w += (size_t)HID * HID * 2;
  __bf16* Qb  = (__bf16*)w; w += (size_t)BS * HID * 2;
  __bf16* Kbf = (__bf16*)w; w += (size_t)BS * HDIM * 2;
  __bf16* Vbf = (__bf16*)w; w += (size_t)BS * HDIM * 2;
  __bf16* Ob  = (__bf16*)w; w += (size_t)BS * HID * 2;
  unsigned* Mb = (unsigned*)w;  // B*S*S/32 words = 1 MB

  auto cvt = [&](const float* s, __bf16* d, int n) {
    f32_to_bf16_kernel<<<dim3((n / 8 + 255) / 256), 256, 0, stream>>>(s, d, n);
  };
  cvt(X,  Xbf, BS * HID);
  cvt(Wq, Wqb, HID * HID);
  cvt(Wk, Wkb, HID * HDIM);
  cvt(Wv, Wvb, HID * HDIM);
  cvt(Wo, Wob, HID * HID);

  pack_mask_kernel<<<dim3((2 * S_DIM * S_DIM) / 256), 256, 0, stream>>>(mk, Mb);

  gemm_bf16_wmma<__bf16><<<dim3(HID / 128, BS / 256), 256, 0, stream>>>(
      Xbf, Wqb, bq, Qb, BS, HID, HID);
  gemm_bf16_wmma<__bf16><<<dim3(HDIM / 128, BS / 256), 256, 0, stream>>>(
      Xbf, Wkb, bk, Kbf, BS, HDIM, HID);
  gemm_bf16_wmma<__bf16><<<dim3(HDIM / 128, BS / 256), 256, 0, stream>>>(
      Xbf, Wvb, bv, Vbf, BS, HDIM, HID);

  mqa_flash_attn<<<dim3(S_DIM / 128, NHEAD, 2), 256, 0, stream>>>(
      Qb, Kbf, Vbf, Mb, Ob);

  gemm_bf16_wmma<float><<<dim3(HID / 128, BS / 256), 256, 0, stream>>>(
      Ob, Wob, bo, out, BS, HID, HID);
}
